// SwinLSTMCellWithFusion_46179488366724
// MI455X (gfx1250) — matrix operand order repaired and checked by reading, not compiled, not run on hardware
//
#include <hip/hip_runtime.h>
#include <cstdint>

// SwinLSTM fused cell for MI455X (gfx1250, wave32).
// All GEMMs use v_wmma_f32_16x16x32_f16 (f16 in, f32 accumulate), with
// weights pre-converted once per launch into transposed f16 (N x K) so the
// B operand is staged with pure b128 traffic. K-loops fully unrolled via
// templates; global_prefetch_b8 issued for the next A tile. Attention keeps
// rel-pos bias and shift-mask codes in LDS.
// Shapes: B=8, H=W=64, L=4096, C=128, NH=4, HD=32, WIN=8, N=64, DEPTH=2.

#define B_   8
#define H__  64
#define W__  64
#define C_   128
#define L_   (H__ * W__)      // 4096
#define M_   (B_ * L_)        // 32768 tokens
#define NH_  4
#define HD_  32

typedef _Float16 v16h __attribute__((ext_vector_type(16)));
typedef float    v8f  __attribute__((ext_vector_type(8)));

__device__ __forceinline__ float warp_sum(float v) {
#pragma unroll
  for (int o = 16; o > 0; o >>= 1) v += __shfl_xor(v, o, 32);
  return v;
}

__device__ __forceinline__ int mask_region(int x) {
  // slices for H=64, WIN=8, SHIFT=4: [0,56) [56,60) [60,64)
  return x < 56 ? 0 : (x < 60 ? 1 : 2);
}

// Weight convert+transpose: out[n*K + k] = (f16) W[k*N + n]
__global__ void convw_k(const float* __restrict__ W, _Float16* __restrict__ out,
                        int K, int N)
{
  const int i = blockIdx.x * 256 + threadIdx.x;
  if (i >= K * N) return;
  const int k = i / N, n = i % N;
  out[(size_t)n * K + k] = (_Float16)W[i];
}

// ---------------------------------------------------------------------------
// f16-WMMA GEMM: out[M,N] = epilogue( A[M,K] @ W[K,N] + bias[N] )
// Wt: transposed f16 weights (N x K). A = concat(A1,A2) at 'ksplit'.
// Block: 256 thr = 8 waves; tile 64(M) x 64(N); K-step 64 (fully unrolled).
// Wave: 32(M) x 16(N) -> two accumulators, 4 WMMAs per barrier pair.
// MODE 0: +bias   1: gelu(+bias)   2: res + (+bias)   3: relu((+bias)*res)
// ---------------------------------------------------------------------------
template <int K, int MODE>
__global__ __launch_bounds__(256) void gemm_wmma(
    const float* __restrict__ A1, const float* __restrict__ A2, int ksplit,
    const _Float16* __restrict__ Wt, const float* __restrict__ bias,
    const float* __restrict__ res, float* __restrict__ out, int N)
{
  __shared__ _Float16 As[64][72];   // 64 x 64 A tile, rows padded to 144B
  __shared__ _Float16 Bs[64][72];   // 64(N) x 64(K) from transposed weights

  const int tid  = threadIdx.x;
  const int m0   = blockIdx.x * 64;
  const int n0   = blockIdx.y * 64;
  const int wave = tid >> 5, lane = tid & 31;
  const int mw   = (wave & 1) * 32;        // wave's M offset (2 subtiles)
  const int nw   = (wave >> 1) * 16;       // wave's N offset
  const int ml   = lane & 15, hi = lane >> 4;
  const int k0   = hi * 8;                 // A frag chunk base within 32-k
  const int kb   = hi * 16;                // unused; B frags use k0 like A
  const int s2   = K - ksplit;

  // staging roles
  const int ar = tid >> 2;                 // row 0..63
  const int ac = (tid & 3) * 16;           // col 0,16,32,48

  v8f c0 = {}, c1 = {};
#pragma unroll
  for (int kk = 0; kk < K; kk += 64) {
    { // stage A: 16 halfs/thread as two b128 stores
      const int gm = m0 + ar;
#pragma unroll
      for (int half = 0; half < 2; ++half) {
        const int kbase = kk + ac + half * 8;   // aligned 8, never crosses ksplit
        const float* src;
        if (kbase < ksplit) src = &A1[(size_t)gm * ksplit + kbase];
        else                src = &A2[(size_t)gm * s2 + (kbase - ksplit)];
        const float4 f0 = *reinterpret_cast<const float4*>(src);
        const float4 f1 = *reinterpret_cast<const float4*>(src + 4);
        _Float16 t8[8] __attribute__((aligned(16)));
        t8[0] = (_Float16)f0.x; t8[1] = (_Float16)f0.y;
        t8[2] = (_Float16)f0.z; t8[3] = (_Float16)f0.w;
        t8[4] = (_Float16)f1.x; t8[5] = (_Float16)f1.y;
        t8[6] = (_Float16)f1.z; t8[7] = (_Float16)f1.w;
        *reinterpret_cast<uint4*>(&As[ar][ac + half * 8]) =
            *reinterpret_cast<const uint4*>(t8);
      }
      // prefetch next A tile rows into cache while we compute
      if (kk + 64 < K) {
        const int kn = kk + 64 + ac;
        const float* pf = (kn < ksplit) ? &A1[(size_t)gm * ksplit + kn]
                                        : &A2[(size_t)gm * s2 + (kn - ksplit)];
        __builtin_prefetch(pf, 0, 1);
      }
    }
    { // stage B: contiguous f16 rows of transposed weights, b128 in/out
      const uint4 w0 = *reinterpret_cast<const uint4*>(
          &Wt[(size_t)(n0 + ar) * K + kk + ac]);
      const uint4 w1 = *reinterpret_cast<const uint4*>(
          &Wt[(size_t)(n0 + ar) * K + kk + ac + 8]);
      *reinterpret_cast<uint4*>(&Bs[ar][ac])     = w0;
      *reinterpret_cast<uint4*>(&Bs[ar][ac + 8]) = w1;
    }
    __syncthreads();
#pragma unroll
    for (int kh = 0; kh < 64; kh += 32) {
      v16h a0, a1, b;
      uint4* bp = reinterpret_cast<uint4*>(&b);
      bp[0] = *reinterpret_cast<const uint4*>(&Bs[nw + ml][kh + k0]);
      bp[1] = *reinterpret_cast<const uint4*>(&Bs[nw + ml][kh + k0 + 16]);
      uint4* ap0 = reinterpret_cast<uint4*>(&a0);
      ap0[0] = *reinterpret_cast<const uint4*>(&As[mw + ml][kh + k0]);
      ap0[1] = *reinterpret_cast<const uint4*>(&As[mw + ml][kh + k0 + 16]);
      uint4* ap1 = reinterpret_cast<uint4*>(&a1);
      ap1[0] = *reinterpret_cast<const uint4*>(&As[mw + 16 + ml][kh + k0]);
      ap1[1] = *reinterpret_cast<const uint4*>(&As[mw + 16 + ml][kh + k0 + 16]);
      c0 = __builtin_amdgcn_wmma_f32_16x16x32_f16(false, a0, false, b,
                                                  (short)0, c0, false, false);
      c1 = __builtin_amdgcn_wmma_f32_16x16x32_f16(false, a1, false, b,
                                                  (short)0, c1, false, false);
    }
    __syncthreads();
  }

  const int nn = n0 + nw + ml;
  const float bv = bias[nn];
#pragma unroll
  for (int sub = 0; sub < 2; ++sub) {
    const v8f& c = sub ? c1 : c0;
#pragma unroll
    for (int r = 0; r < 8; ++r) {
      const int mm = m0 + mw + sub * 16 + hi * 8 + r;
      const size_t idx = (size_t)mm * N + nn;
      float v = c[r] + bv;
      if (MODE == 1) {
        v = 0.5f * v * (1.0f + erff(v * 0.7071067811865475f));  // exact GELU
      } else if (MODE == 2) {
        v = res[idx] + v;                                       // residual
      } else if (MODE == 3) {
        v = v * res[idx];                                       // * then relu
        v = v > 0.f ? v : 0.f;
      }
      out[idx] = v;
    }
  }
}

// ---------------------------------------------------------------------------
// LayerNorm fused with cyclic shift + window partition (one wave per token).
// ---------------------------------------------------------------------------
__global__ __launch_bounds__(256) void ln_window_k(
    const float* __restrict__ x, const float* __restrict__ gamma,
    const float* __restrict__ beta, float* __restrict__ xw, int shift)
{
  const int wave = threadIdx.x >> 5, lane = threadIdx.x & 31;
  const int tok  = blockIdx.x * 8 + wave;         // windowed token id
  const int wall = tok >> 6;                      // b*64 + window
  const int b    = wall >> 6;
  const int wi   = wall & 63;
  const int n    = tok & 63;
  const int oh   = ((wi >> 3) << 3) + (n >> 3);
  const int ow   = ((wi & 7) << 3) + (n & 7);
  const int sh   = (oh + shift) & 63;             // inverse of roll(-shift)
  const int sw   = (ow + shift) & 63;
  const size_t src = ((size_t)b * L_ + sh * 64 + sw) * C_;

  float v[4], s = 0.f, sq = 0.f;
#pragma unroll
  for (int j = 0; j < 4; ++j) {
    v[j] = x[src + lane * 4 + j];
    s += v[j];
    sq += v[j] * v[j];
  }
  s  = warp_sum(s);
  sq = warp_sum(sq);
  const float mean = s * (1.f / 128.f);
  const float rstd = rsqrtf(sq * (1.f / 128.f) - mean * mean + 1e-5f);
  const size_t dst = (size_t)tok * C_;
#pragma unroll
  for (int j = 0; j < 4; ++j) {
    const int cc = lane * 4 + j;
    xw[dst + cc] = (v[j] - mean) * rstd * gamma[cc] + beta[cc];
  }
}

__global__ __launch_bounds__(256) void ln_plain_k(
    const float* __restrict__ x, const float* __restrict__ gamma,
    const float* __restrict__ beta, float* __restrict__ out)
{
  const int wave = threadIdx.x >> 5, lane = threadIdx.x & 31;
  const int tok  = blockIdx.x * 8 + wave;
  const size_t src = (size_t)tok * C_;
  float v[4], s = 0.f, sq = 0.f;
#pragma unroll
  for (int j = 0; j < 4; ++j) {
    v[j] = x[src + lane * 4 + j];
    s += v[j];
    sq += v[j] * v[j];
  }
  s  = warp_sum(s);
  sq = warp_sum(sq);
  const float mean = s * (1.f / 128.f);
  const float rstd = rsqrtf(sq * (1.f / 128.f) - mean * mean + 1e-5f);
#pragma unroll
  for (int j = 0; j < 4; ++j) {
    const int cc = lane * 4 + j;
    out[src + cc] = (v[j] - mean) * rstd * gamma[cc] + beta[cc];
  }
}

// ---------------------------------------------------------------------------
// Window attention, one block per (window, head). 256 thr = 8 waves.
// QK^T: 16 tiles, one 16x16x32 WMMA each (K = HD = 32).
// Rel-pos bias slice (225 floats) and mask codes (64) cached in LDS.
// Softmax: 4 threads per row via LDS partial reduce. PV: 8 tiles x 2 K-steps.
// ---------------------------------------------------------------------------
__global__ __launch_bounds__(256) void attn_k(
    const float* __restrict__ qkv, const float* __restrict__ rp,
    float* __restrict__ aout, int shift)
{
  __shared__ _Float16 qs[64][40];   // token x d   (q, pre-scaled)
  __shared__ _Float16 ks_[64][40];  // token x d   (Bt for QK^T)
  __shared__ _Float16 vt[32][72];   // d x token   (Bt for PV)
  __shared__ float    sc[64][64];   // scores
  __shared__ _Float16 ps[64][72];   // probs, row-major (A operand)
  __shared__ float    rps[225];     // rel-pos bias, this head
  __shared__ int      cnt[64];      // shift-mask region code per token
  __shared__ float    red4[64][4];  // softmax partials

  const int tid  = threadIdx.x;
  const int wblk = blockIdx.x >> 2;   // 0..511 (b*64 + window)
  const int h    = blockIdx.x & 3;
  const int wi   = wblk & 63;         // window position (for shift mask)
  const int wave = tid >> 5, lane = tid & 31;
  const int ml   = lane & 15, hi2 = lane >> 4;
  const int k0   = hi2 * 8, kb = hi2 * 16;

  // cache rel-pos bias slice + mask region codes
  if (tid < 225) rps[tid] = rp[tid * NH_ + h];
  if (tid < 64) {
    if (shift > 0) {
      const int ph = ((wi >> 3) << 3) + (tid >> 3);
      const int pw = ((wi & 7) << 3) + (tid & 7);
      cnt[tid] = mask_region(ph) * 3 + mask_region(pw);
    } else {
      cnt[tid] = 0;
    }
  }

  { // load q,k,v; q/k as packed b128 LDS stores, v scattered (transpose)
    const int t  = tid >> 2;
    const int d0 = (tid & 3) * 8;
    const size_t base = ((size_t)wblk * 64 + t) * 384 + h * HD_ + d0;
    _Float16 tq[8] __attribute__((aligned(16)));
    _Float16 tk[8] __attribute__((aligned(16)));
#pragma unroll
    for (int j = 0; j < 8; ++j) {
      tq[j] = (_Float16)(qkv[base + j] * 0.17677669529663687f); // HD^-0.5
      tk[j] = (_Float16)qkv[base + 128 + j];
      vt[d0 + j][t] = (_Float16)qkv[base + 256 + j];
    }
    *reinterpret_cast<uint4*>(&qs[t][d0])  = *reinterpret_cast<const uint4*>(tq);
    *reinterpret_cast<uint4*>(&ks_[t][d0]) = *reinterpret_cast<const uint4*>(tk);
  }
  __syncthreads();

  // ---- scores = q @ k^T + bias (+ mask) ----
#pragma unroll
  for (int tt = 0; tt < 2; ++tt) {
    const int tl = wave + tt * 8;
    const int it = tl >> 2, jt = tl & 3;
    v16h a, b;
    v8f  c = {};
    uint4* ap = reinterpret_cast<uint4*>(&a);
    ap[0] = *reinterpret_cast<const uint4*>(&qs[it * 16 + ml][k0]);
    ap[1] = *reinterpret_cast<const uint4*>(&qs[it * 16 + ml][k0 + 16]);
    uint4* bp = reinterpret_cast<uint4*>(&b);
    bp[0] = *reinterpret_cast<const uint4*>(&ks_[jt * 16 + ml][k0]);
    bp[1] = *reinterpret_cast<const uint4*>(&ks_[jt * 16 + ml][k0 + 16]);
    c = __builtin_amdgcn_wmma_f32_16x16x32_f16(false, a, false, b,
                                               (short)0, c, false, false);
    const int km = jt * 16 + ml;            // key token (constant over r)
    const int ck = cnt[km];
#pragma unroll
    for (int r = 0; r < 8; ++r) {
      const int qn = it * 16 + hi2 * 8 + r; // query token
      const int d0i = (qn >> 3) - (km >> 3) + 7;
      const int d1i = (qn & 7) - (km & 7) + 7;
      const float bv = rps[d0i * 15 + d1i];
      const float mv = (cnt[qn] != ck) ? -100.f : 0.f;
      sc[qn][km] = c[r] + bv + mv;
    }
  }
  __syncthreads();

  // ---- softmax: 4 threads per row ----
  {
    const int row = tid >> 2, q4 = (tid & 3) * 16;
    float mx = -1e30f;
#pragma unroll
    for (int j = 0; j < 16; ++j) mx = fmaxf(mx, sc[row][q4 + j]);
    red4[row][tid & 3] = mx;
    __syncthreads();
    mx = fmaxf(fmaxf(red4[row][0], red4[row][1]),
               fmaxf(red4[row][2], red4[row][3]));
    float s = 0.f;
#pragma unroll
    for (int j = 0; j < 16; ++j) s += expf(sc[row][q4 + j] - mx);
    __syncthreads();
    red4[row][tid & 3] = s;
    __syncthreads();
    const float inv = 1.f / (red4[row][0] + red4[row][1] +
                             red4[row][2] + red4[row][3]);
#pragma unroll
    for (int j = 0; j < 16; ++j)
      ps[row][q4 + j] = (_Float16)(expf(sc[row][q4 + j] - mx) * inv);
  }
  __syncthreads();

  // ---- out = probs @ v  (64x32, K=64 in two steps) ----
  {
    const int it = wave & 3, jt = wave >> 2;   // 4 x 2 tiles
    v8f c = {};
#pragma unroll
    for (int ks2 = 0; ks2 < 2; ++ks2) {
      v16h a, b;
      uint4* ap = reinterpret_cast<uint4*>(&a);
      ap[0] = *reinterpret_cast<const uint4*>(&ps[it * 16 + ml][ks2 * 32 + k0]);
      ap[1] = *reinterpret_cast<const uint4*>(&ps[it * 16 + ml][ks2 * 32 + k0 + 16]);
      uint4* bp = reinterpret_cast<uint4*>(&b);
      bp[0] = *reinterpret_cast<const uint4*>(&vt[jt * 16 + ml][ks2 * 32 + k0]);
      bp[1] = *reinterpret_cast<const uint4*>(&vt[jt * 16 + ml][ks2 * 32 + k0 + 16]);
      c = __builtin_amdgcn_wmma_f32_16x16x32_f16(false, a, false, b,
                                                 (short)0, c, false, false);
    }
#pragma unroll
    for (int r = 0; r < 8; ++r) {
      const int t = it * 16 + hi2 * 8 + r;
      const int d = jt * 16 + ml;
      aout[((size_t)wblk * 64 + t) * C_ + h * HD_ + d] = c[r];
    }
  }
}

// window_reverse + roll(+shift) + residual add (in place on x)
__global__ void permute_add_k(float* __restrict__ x,
                              const float* __restrict__ ptmp, int shift)
{
  const int i  = blockIdx.x * blockDim.x + threadIdx.x;  // over B*L*C
  const int cc = i & 127;
  const int l  = (i >> 7) & 4095;
  const int b  = i >> 19;
  const int hh = l >> 6, ww = l & 63;
  const int h2 = (hh - shift) & 63, w2 = (ww - shift) & 63;
  const int wiw = ((h2 >> 3) << 3) + (w2 >> 3);
  const int n   = ((h2 & 7) << 3) + (w2 & 7);
  x[i] += ptmp[(((size_t)b * 64 + wiw) * 64 + n) * C_ + cc];
}

// ---------------------------------------------------------------------------
// Deformable 1D conv branch. One 128-thread block per (b,l) token.
// Output written in (B, C, L) order -> later reinterpreted as (B, L, C)
// (faithful to the reference's raw memory reshape).
// ---------------------------------------------------------------------------
__global__ __launch_bounds__(128) void local_k(
    const float* __restrict__ xt,
    const float* __restrict__ off_w, const float* __restrict__ off_b,
    const float* __restrict__ msk_w, const float* __restrict__ msk_b,
    float* __restrict__ outb)
{
  __shared__ float red[4][6];
  __shared__ float offs[3], msks[3];
  const int tid = threadIdx.x;                 // channel
  const int b   = blockIdx.x >> 12;
  const int l   = blockIdx.x & 4095;
  const size_t xb = (size_t)b * L_ * C_;

  const float xm = (l > 0)      ? xt[xb + (size_t)(l - 1) * C_ + tid] : 0.f;
  const float x0 = xt[xb + (size_t)l * C_ + tid];
  const float xp = (l < L_ - 1) ? xt[xb + (size_t)(l + 1) * C_ + tid] : 0.f;

  float po[3], pm[3];
#pragma unroll
  for (int k = 0; k < 3; ++k) {
    const float* ow = &off_w[(k * C_ + tid) * 3];
    const float* mw = &msk_w[(k * C_ + tid) * 3];
    po[k] = xm * ow[0] + x0 * ow[1] + xp * ow[2];
    pm[k] = xm * mw[0] + x0 * mw[1] + xp * mw[2];
  }
  const int wave = tid >> 5, lane = tid & 31;
#pragma unroll
  for (int k = 0; k < 3; ++k) {
    po[k] = warp_sum(po[k]);
    pm[k] = warp_sum(pm[k]);
  }
  if (lane == 0) {
#pragma unroll
    for (int k = 0; k < 3; ++k) {
      red[wave][k]     = po[k];
      red[wave][3 + k] = pm[k];
    }
  }
  __syncthreads();
  if (tid < 3) {
    const float o = red[0][tid] + red[1][tid] + red[2][tid] + red[3][tid] + off_b[tid];
    const float m = red[0][3 + tid] + red[1][3 + tid] + red[2][3 + tid] +
                    red[3][3 + tid] + msk_b[tid];
    offs[tid] = o;
    msks[tid] = 1.f / (1.f + expf(-m));
  }
  __syncthreads();

  float acc = 0.f;
#pragma unroll
  for (int k = 0; k < 3; ++k) {
    float pos = (float)l + offs[k];
    pos = fminf(fmaxf(pos, 0.f), (float)(L_ - 1));
    const int   fp = (int)floorf(pos);
    const int   cp = min(fp + 1, L_ - 1);
    const float al = pos - (float)fp;
    const float sf = xt[xb + (size_t)fp * C_ + tid];
    const float sc = xt[xb + (size_t)cp * C_ + tid];
    acc += (sf * (1.f - al) + sc * al) * msks[k];
  }
  outb[((size_t)b * C_ + tid) * L_ + l] = acc;   // (B, C, L) order
}

// LSTM-style gating: hy = sig(fu) * tanh(sig(fu)*(cx + tanh(fu)))
__global__ void gate_k(const float* __restrict__ fu,
                       const float* __restrict__ cx, float* __restrict__ out)
{
  const size_t i = (size_t)blockIdx.x * blockDim.x + threadIdx.x;
  const float f    = fu[i];
  const float g    = 1.f / (1.f + expf(-f));
  const float cell = tanhf(f);
  const float cy   = g * (cx[i] + cell);
  out[i] = g * tanhf(cy);
}

// ---------------------------------------------------------------------------
extern "C" void kernel_launch(void* const* d_in, const int* in_sizes, int n_in,
                              void* d_out, int out_size, void* d_ws, size_t ws_size,
                              hipStream_t stream)
{
  const float* xt    = (const float*)d_in[0];
  const float* hx    = (const float*)d_in[1];
  const float* cx    = (const float*)d_in[2];
  const float* red_w = (const float*)d_in[3];
  const float* red_b = (const float*)d_in[4];
  const float* n1g   = (const float*)d_in[5];
  const float* n1b   = (const float*)d_in[6];
  const float* qkvw  = (const float*)d_in[7];
  const float* qkvb  = (const float*)d_in[8];
  const float* rp    = (const float*)d_in[9];
  const float* pw    = (const float*)d_in[10];
  const float* pb    = (const float*)d_in[11];
  const float* n2g   = (const float*)d_in[12];
  const float* n2b   = (const float*)d_in[13];
  const float* f1w   = (const float*)d_in[14];
  const float* f1b   = (const float*)d_in[15];
  const float* f2w   = (const float*)d_in[16];
  const float* f2b   = (const float*)d_in[17];
  const float* off_w = (const float*)d_in[18];
  const float* off_b = (const float*)d_in[19];
  const float* msk_w = (const float*)d_in[20];
  const float* msk_b = (const float*)d_in[21];
  const float* lf_w  = (const float*)d_in[22];
  const float* lf_b  = (const float*)d_in[23];
  const float* gf_w  = (const float*)d_in[24];
  const float* gf_b  = (const float*)d_in[25];
  const float* ff_w  = (const float*)d_in[26];
  const float* ff_b  = (const float*)d_in[27];

  // Workspace: 8 activation slots of B*L*C f32 (128 MB) + f16 weight pool.
  float* ws = (float*)d_ws;
  const size_t SZ = (size_t)M_ * C_;
  float* g      = ws + 0 * SZ;          // persistent global branch
  float* xw     = ws + 1 * SZ;          // windowed/normed acts; later: local
  float* qkv    = ws + 2 * SZ;          // slots 2..4
  float* aout   = ws + 5 * SZ;          // attn out; later: t1 (lg)
  float* ptmp   = ws + 6 * SZ;          // proj out; later: fu_pre
  float* hbuf   = ws + 7 * SZ;          // ln2 out; later: fu
  float* hid    = ws + 2 * SZ;          // MLP hidden, reuses slots 2..5
  float* localb = xw;
  float* t1     = aout;
  float* fupre  = ptmp;
  float* fu     = hbuf;

  // f16 transposed weight pool (~1 MB)
  _Float16* wt = (_Float16*)(ws + 8 * SZ);
  size_t wo = 0;
  _Float16* red_wT  = wt + wo; wo += (size_t)256 * 128;
  _Float16* qkvwT0  = wt + wo; wo += (size_t)128 * 384;
  _Float16* qkvwT1  = wt + wo; wo += (size_t)128 * 384;
  _Float16* pwT0    = wt + wo; wo += (size_t)128 * 128;
  _Float16* pwT1    = wt + wo; wo += (size_t)128 * 128;
  _Float16* f1wT0   = wt + wo; wo += (size_t)128 * 512;
  _Float16* f1wT1   = wt + wo; wo += (size_t)128 * 512;
  _Float16* f2wT0   = wt + wo; wo += (size_t)512 * 128;
  _Float16* f2wT1   = wt + wo; wo += (size_t)512 * 128;
  _Float16* lfT     = wt + wo; wo += (size_t)128 * 128;
  _Float16* gfT     = wt + wo; wo += (size_t)128 * 128;
  _Float16* ffT     = wt + wo; wo += (size_t)128 * 128;

  const dim3 blk(256);
  auto cw = [&](const float* W, _Float16* o, int K, int N) {
    convw_k<<<dim3((K * N + 255) / 256), blk, 0, stream>>>(W, o, K, N);
  };
  cw(red_w, red_wT, 256, 128);
  cw(qkvw,                    qkvwT0, 128, 384);
  cw(qkvw + (size_t)128*384,  qkvwT1, 128, 384);
  cw(pw,                      pwT0,   128, 128);
  cw(pw + (size_t)128*128,    pwT1,   128, 128);
  cw(f1w,                     f1wT0,  128, 512);
  cw(f1w + (size_t)128*512,   f1wT1,  128, 512);
  cw(f2w,                     f2wT0,  512, 128);
  cw(f2w + (size_t)512*128,   f2wT1,  512, 128);
  cw(lf_w, lfT, 128, 128);
  cw(gf_w, gfT, 128, 128);
  cw(ff_w, ffT, 128, 128);

  // g = concat(xt, hx) @ red_w + red_b   (M=32768, K=256, N=128)
  gemm_wmma<256, 0><<<dim3(M_ / 64, 128 / 64), blk, 0, stream>>>(
      xt, hx, 128, red_wT, red_b, nullptr, g, 128);

  const _Float16* qkvT[2] = {qkvwT0, qkvwT1};
  const _Float16* pwT[2]  = {pwT0, pwT1};
  const _Float16* f1T[2]  = {f1wT0, f1wT1};
  const _Float16* f2T[2]  = {f2wT0, f2wT1};
  const int shifts[2] = {0, 4};
  for (int i = 0; i < 2; ++i) {
    const int sh = shifts[i];
    ln_window_k<<<dim3(M_ / 8), blk, 0, stream>>>(
        g, n1g + i * C_, n1b + i * C_, xw, sh);
    gemm_wmma<128, 0><<<dim3(M_ / 64, 384 / 64), blk, 0, stream>>>(
        xw, nullptr, 128, qkvT[i], qkvb + i * 384, nullptr, qkv, 384);
    attn_k<<<dim3(512 * NH_), blk, 0, stream>>>(
        qkv, rp + (size_t)i * 225 * NH_, aout, sh);
    gemm_wmma<128, 0><<<dim3(M_ / 64, 128 / 64), blk, 0, stream>>>(
        aout, nullptr, 128, pwT[i], pb + i * C_, nullptr, ptmp, 128);
    permute_add_k<<<dim3((M_ * C_) / 256), blk, 0, stream>>>(g, ptmp, sh);
    ln_plain_k<<<dim3(M_ / 8), blk, 0, stream>>>(
        g, n2g + i * C_, n2b + i * C_, hbuf);
    gemm_wmma<128, 1><<<dim3(M_ / 64, 512 / 64), blk, 0, stream>>>(
        hbuf, nullptr, 128, f1T[i], f1b + i * 512, nullptr, hid, 512); // GELU
    gemm_wmma<512, 2><<<dim3(M_ / 64, 128 / 64), blk, 0, stream>>>(
        hid, nullptr, 512, f2T[i], f2b + i * C_, g, g, 128);           // +res
  }

  // local branch + fusion
  local_k<<<dim3(M_), dim3(128), 0, stream>>>(
      xt, off_w, off_b, msk_w, msk_b, localb);
  gemm_wmma<128, 0><<<dim3(M_ / 64, 128 / 64), blk, 0, stream>>>(
      localb, nullptr, 128, lfT, lf_b, nullptr, t1, 128);
  gemm_wmma<128, 3><<<dim3(M_ / 64, 128 / 64), blk, 0, stream>>>(
      g, nullptr, 128, gfT, gf_b, t1, fupre, 128);        // relu(lg*gg)
  gemm_wmma<128, 0><<<dim3(M_ / 64, 128 / 64), blk, 0, stream>>>(
      fupre, nullptr, 128, ffT, ff_b, nullptr, fu, 128);
  gate_k<<<dim3((M_ * C_) / 256), blk, 0, stream>>>(fu, cx, (float*)d_out);
}
